// TransformerEncoder_40707700032313
// MI455X (gfx1250) — matrix operand order repaired and checked by reading, compiled, not run on hardware
//
#include <hip/hip_runtime.h>
#include <cstdint>

typedef __attribute__((ext_vector_type(16))) __bf16 v16bf;
typedef __attribute__((ext_vector_type(8)))  float  v8f;

#define DIM     1024
#define SEQ     1024
#define BATCH   8
#define NHEADS  16
#define HEADDIM 64
#define MLPDIM  4096
#define NROWS   (BATCH * SEQ)   // 8192
#define DEPTHL  6
#define EPSLN   1e-5f
#define ATTSCALE 0.03125f       // DIM^-0.5 (module scales by full dim)

union Frag { v16bf v; uint4 u[2]; };
union B8   { uint4 u; __bf16 e[8]; };
union F8   { float4 f[2]; float s[8]; };

static __device__ inline v8f wmma_bf16(v16bf a, v16bf b, v8f c) {
  return __builtin_amdgcn_wmma_f32_16x16x32_bf16(false, a, false, b, (short)0, c, false, false);
}

// ---------------------------------------------------------------- fp32 [K][N] -> bf16 transposed [N][K]
__global__ __launch_bounds__(256)
void cvt_t_bf16(const float* __restrict__ src, __bf16* __restrict__ dst, int K, int N) {
  __shared__ float tile[32][33];
  int n0 = blockIdx.x * 32, k0 = blockIdx.y * 32;
  int tx = threadIdx.x, ty = threadIdx.y;             // 32 x 8
#pragma unroll
  for (int j = 0; j < 32; j += 8)
    tile[ty + j][tx] = src[(size_t)(k0 + ty + j) * N + n0 + tx];   // tile[k][n]
  __syncthreads();
#pragma unroll
  for (int j = 0; j < 32; j += 8)
    dst[(size_t)(n0 + ty + j) * K + k0 + tx] = (__bf16)tile[tx][ty + j];
}

// ---------------------------------------------------------------- LayerNorm (one row per block)
__global__ __launch_bounds__(256)
void layernorm_k(const float* __restrict__ x, const float* __restrict__ gamma,
                 const float* __restrict__ beta, __bf16* __restrict__ out) {
  __shared__ float red[256], red2[256];
  int row = blockIdx.x;
  int t = threadIdx.x;
  const float* xr = x + (size_t)row * DIM;
  float4 vv = *(const float4*)(xr + t * 4);
  float s  = vv.x + vv.y + vv.z + vv.w;
  float s2 = vv.x * vv.x + vv.y * vv.y + vv.z * vv.z + vv.w * vv.w;
  red[t] = s; red2[t] = s2;
  __syncthreads();
  for (int off = 128; off > 0; off >>= 1) {
    if (t < off) { red[t] += red[t + off]; red2[t] += red2[t + off]; }
    __syncthreads();
  }
  float mu  = red[0] * (1.0f / DIM);
  float var = red2[0] * (1.0f / DIM) - mu * mu;
  float rs  = rsqrtf(var + EPSLN);
  __bf16* orow = out + (size_t)row * DIM;
  const float* vp = (const float*)&vv;
#pragma unroll
  for (int j = 0; j < 4; ++j) {
    int c = t * 4 + j;
    float val = (vp[j] - mu) * rs * gamma[c] + beta[c];
    orow[c] = (__bf16)val;
  }
}

// ---------------------------------------------------------------- GEMM (transposed-operand form, no LDS scatter)
// out[row, col] = sum_k act[row,k] * Wt[col,k] + bias[col]
// WMMA: A = Wt tile (M dim = out col), B = act tile (N dim = act row).
// Block: 128 cols x 128 rows, BK=32, double-buffered LDS, 8 waves each 32 cols x 64 rows.
// EPI 0: store bf16   EPI 1: exact GELU, store bf16   EPI 2: +resid, store f32
template <int EPI>
__global__ __launch_bounds__(256)
void gemm_k(const __bf16* __restrict__ act, const __bf16* __restrict__ wt,
            const float* __restrict__ bias, void* __restrict__ outp,
            const float* __restrict__ resid, int M, int Nf, int K) {
  (void)M;
  __shared__ __attribute__((aligned(16))) __bf16 Ws[2][128][40];   // [col][k]
  __shared__ __attribute__((aligned(16))) __bf16 Xs[2][128][40];   // [row][k]
  int tid = threadIdx.x;
  int lane = tid & 31, wave = tid >> 5;
  int wm = wave >> 1, wn = wave & 1;           // wm: col group (0..3), wn: row group (0..1)
  int cblk = blockIdx.x * 128, mblk = blockIdx.y * 128;
  int arow  = lane & 15;
  int khalf = lane >> 4;
  int c0    = (lane < 16) ? 0 : 8;

  const v8f vz = {0, 0, 0, 0, 0, 0, 0, 0};
  v8f acc[2][4];
#pragma unroll
  for (int i = 0; i < 2; ++i)
#pragma unroll
    for (int j = 0; j < 4; ++j) acc[i][j] = vz;

  auto loadTiles = [&](int kt, int buf) {
    int k0 = kt * 32;
#pragma unroll
    for (int p = 0; p < 2; ++p) {
      int idx = p * 256 + tid;
      int r = idx >> 2, c = (idx & 3) * 8;
      *(uint4*)&Ws[buf][r][c] = *(const uint4*)(wt  + (size_t)(cblk + r) * K + k0 + c);
      *(uint4*)&Xs[buf][r][c] = *(const uint4*)(act + (size_t)(mblk + r) * K + k0 + c);
    }
  };

  int nkt = K >> 5;
  loadTiles(0, 0);
  __syncthreads();

  for (int kt = 0; kt < nkt; ++kt) {
    int cur = kt & 1;
    if (kt + 1 < nkt) loadTiles(kt + 1, cur ^ 1);

    Frag af[2], bfr[4];
#pragma unroll
    for (int i = 0; i < 2; ++i) {                       // A: weight cols
      const __bf16* p0 = &Ws[cur][wm * 32 + i * 16 + arow][c0];
      af[i].u[0] = *(const uint4*)p0;
      af[i].u[1] = *(const uint4*)(p0 + 16);
    }
#pragma unroll
    for (int j = 0; j < 4; ++j) {                       // B: activation rows
      const __bf16* p0 = &Xs[cur][wn * 64 + j * 16 + arow][khalf * 8];
      bfr[j].u[0] = *(const uint4*)p0;
      bfr[j].u[1] = *(const uint4*)(p0 + 16);
    }
#pragma unroll
    for (int i = 0; i < 2; ++i)
#pragma unroll
      for (int j = 0; j < 4; ++j)
        acc[i][j] = wmma_bf16(af[i].v, bfr[j].v, acc[i][j]);
    __syncthreads();
  }

  // Epilogue: lane owns 8 contiguous output columns for one activation row.
#pragma unroll
  for (int i = 0; i < 2; ++i)
#pragma unroll
    for (int j = 0; j < 4; ++j) {
      int row  = mblk + wn * 64 + j * 16 + arow;        // N position = act row
      int colb = cblk + wm * 32 + i * 16 + 8 * khalf;   // M base = out col (8 contiguous)
      F8 bia;
      bia.f[0] = *(const float4*)(bias + colb);
      bia.f[1] = *(const float4*)(bias + colb + 4);
      if (EPI == 2) {
        const float* rp = resid + (size_t)row * Nf + colb;
        float4 r0 = *(const float4*)rp;
        float4 r1 = *(const float4*)(rp + 4);
        F8 ov;
#pragma unroll
        for (int r = 0; r < 8; ++r) ov.s[r] = acc[i][j][r] + bia.s[r];
        ov.f[0].x += r0.x; ov.f[0].y += r0.y; ov.f[0].z += r0.z; ov.f[0].w += r0.w;
        ov.f[1].x += r1.x; ov.f[1].y += r1.y; ov.f[1].z += r1.z; ov.f[1].w += r1.w;
        float* op = (float*)outp + (size_t)row * Nf + colb;
        *(float4*)op = ov.f[0];
        *(float4*)(op + 4) = ov.f[1];
      } else {
        B8 ob;
#pragma unroll
        for (int r = 0; r < 8; ++r) {
          float val = acc[i][j][r] + bia.s[r];
          if (EPI == 1) val = 0.5f * val * (1.0f + erff(val * 0.70710678118654752f));
          ob.e[r] = (__bf16)val;
        }
        *(uint4*)((__bf16*)outp + (size_t)row * Nf + colb) = ob.u;
      }
    }
}

// ---------------------------------------------------------------- Flash attention, x += softmax(QK^T*s)V
__global__ __launch_bounds__(256)
void attn_k(const __bf16* __restrict__ q, const __bf16* __restrict__ kmat,
            const __bf16* __restrict__ v, float* __restrict__ x) {
  __shared__ __attribute__((aligned(16))) __bf16 Qs[64][72];
  __shared__ __attribute__((aligned(16))) __bf16 Ks[64][72];
  __shared__ __attribute__((aligned(16))) __bf16 Vs[64][72];   // Vs[d][key]
  __shared__ __attribute__((aligned(16))) __bf16 Ps[64][72];
  __shared__ float Sb[64][66];
  __shared__ float mrow[64], lrow[64], fac[64];

  int tid = threadIdx.x;
  int lane = tid & 31, wave = tid >> 5;
  int wm = wave >> 1, wn = wave & 1;
  int arow = lane & 15, khalf = lane >> 4;
  int c0 = (lane < 16) ? 0 : 8;
  int qt = blockIdx.x, h = blockIdx.y, b = blockIdx.z;

  if (tid < 64) { mrow[tid] = -3.0e38f; lrow[tid] = 0.0f; }

  const __bf16* qbase = q + ((size_t)b * SEQ + (size_t)qt * 64) * DIM + h * HEADDIM;
#pragma unroll
  for (int p = 0; p < 2; ++p) {
    int r = p * 32 + (tid >> 3), c = (tid & 7) * 8;
    *(uint4*)&Qs[r][c] = *(const uint4*)(qbase + (size_t)r * DIM + c);
  }
  const v8f vz = {0, 0, 0, 0, 0, 0, 0, 0};
  v8f o[2]; o[0] = vz; o[1] = vz;
  __syncthreads();

  for (int kt = 0; kt < SEQ / 64; ++kt) {
    const __bf16* kbase = kmat + ((size_t)b * SEQ + (size_t)kt * 64) * DIM + h * HEADDIM;
    const __bf16* vbase = v    + ((size_t)b * SEQ + (size_t)kt * 64) * DIM + h * HEADDIM;
#pragma unroll
    for (int p = 0; p < 2; ++p) {
      int r = p * 32 + (tid >> 3), c = (tid & 7) * 8;
      *(uint4*)&Ks[r][c] = *(const uint4*)(kbase + (size_t)r * DIM + c);
      B8 tmp; tmp.u = *(const uint4*)(vbase + (size_t)r * DIM + c);
#pragma unroll
      for (int j = 0; j < 8; ++j) Vs[c + j][r] = tmp.e[j];   // transpose V
    }
    __syncthreads();

    // S = Q K^T (contraction over d=64 -> two K=32 WMMA steps)
    v8f s[2]; s[0] = vz; s[1] = vz;
#pragma unroll
    for (int kk = 0; kk < 2; ++kk) {
      Frag a;
      const __bf16* ap = &Qs[wm * 16 + arow][kk * 32 + c0];
      a.u[0] = *(const uint4*)ap; a.u[1] = *(const uint4*)(ap + 16);
#pragma unroll
      for (int j = 0; j < 2; ++j) {
        Frag bb;
        const __bf16* bp = &Ks[wn * 32 + j * 16 + arow][kk * 32 + khalf * 8];
        bb.u[0] = *(const uint4*)bp; bb.u[1] = *(const uint4*)(bp + 16);
        s[j] = wmma_bf16(a.v, bb.v, s[j]);
      }
    }
#pragma unroll
    for (int j = 0; j < 2; ++j)
#pragma unroll
      for (int r = 0; r < 8; ++r)
        Sb[wm * 16 + r + 8 * khalf][wn * 32 + j * 16 + arow] = s[j][r] * ATTSCALE;
    __syncthreads();

    // online softmax per row
    if (tid < 64) {
      int row = tid;
      float mx = mrow[row];
      float tm = -3.0e38f;
      for (int j = 0; j < 64; ++j) tm = fmaxf(tm, Sb[row][j]);
      float mn = fmaxf(mx, tm);
      float f = __expf(mx - mn);
      float sum = 0.0f;
      for (int j = 0; j < 64; ++j) {
        float e = __expf(Sb[row][j] - mn);
        Ps[row][j] = (__bf16)e;
        sum += e;
      }
      lrow[row] = lrow[row] * f + sum;
      mrow[row] = mn;
      fac[row] = f;
    }
    __syncthreads();

    // rescale O, then O += P V
#pragma unroll
    for (int t = 0; t < 2; ++t)
#pragma unroll
      for (int r = 0; r < 8; ++r)
        o[t][r] *= fac[wm * 16 + r + 8 * khalf];
#pragma unroll
    for (int kk = 0; kk < 2; ++kk) {
      Frag a;
      const __bf16* ap = &Ps[wm * 16 + arow][kk * 32 + c0];
      a.u[0] = *(const uint4*)ap; a.u[1] = *(const uint4*)(ap + 16);
#pragma unroll
      for (int t = 0; t < 2; ++t) {
        Frag bb;
        const __bf16* bp = &Vs[wn * 32 + t * 16 + arow][kk * 32 + khalf * 8];
        bb.u[0] = *(const uint4*)bp; bb.u[1] = *(const uint4*)(bp + 16);
        o[t] = wmma_bf16(a.v, bb.v, o[t]);
      }
    }
    __syncthreads();
  }

  float* xbase = x + ((size_t)b * SEQ + (size_t)qt * 64) * DIM + h * HEADDIM;
#pragma unroll
  for (int t = 0; t < 2; ++t)
#pragma unroll
    for (int r = 0; r < 8; ++r) {
      int row = wm * 16 + r + 8 * khalf;
      int col = wn * 32 + t * 16 + arow;
      xbase[(size_t)row * DIM + col] += o[t][r] * (1.0f / lrow[row]);
    }
}

// ---------------------------------------------------------------- host
extern "C" void kernel_launch(void* const* d_in, const int* in_sizes, int n_in,
                              void* d_out, int out_size, void* d_ws, size_t ws_size,
                              hipStream_t stream) {
  (void)in_sizes; (void)n_in; (void)out_size; (void)ws_size;
  const float* x_in  = (const float*)d_in[0];
  const float* Wq    = (const float*)d_in[1];
  const float* bq    = (const float*)d_in[2];
  const float* Wk    = (const float*)d_in[3];
  const float* bk    = (const float*)d_in[4];
  const float* Wv    = (const float*)d_in[5];
  const float* bv    = (const float*)d_in[6];
  const float* W1    = (const float*)d_in[7];
  const float* b1    = (const float*)d_in[8];
  const float* W2    = (const float*)d_in[9];
  const float* b2    = (const float*)d_in[10];
  const float* gamma = (const float*)d_in[11];
  const float* beta  = (const float*)d_in[12];

  float* x = (float*)d_out;  // residual stream lives in the output buffer

  char* ws = (char*)d_ws;
  size_t off = 0;
  __bf16* Wq_t = (__bf16*)(ws + off); off += (size_t)DIM * DIM * 2;      // [DIM(out)][DIM(in)]
  __bf16* Wk_t = (__bf16*)(ws + off); off += (size_t)DIM * DIM * 2;
  __bf16* Wv_t = (__bf16*)(ws + off); off += (size_t)DIM * DIM * 2;
  __bf16* W1_t = (__bf16*)(ws + off); off += (size_t)DIM * MLPDIM * 2;   // [MLPDIM][DIM]
  __bf16* W2_t = (__bf16*)(ws + off); off += (size_t)MLPDIM * DIM * 2;   // [DIM][MLPDIM]
  __bf16* xn   = (__bf16*)(ws + off); off += (size_t)NROWS * DIM * 2;
  char*   big  = ws + off;                       // 64 MB region, dual use
  __bf16* qb = (__bf16*)(big);
  __bf16* kb = (__bf16*)(big + (size_t)NROWS * DIM * 2);
  __bf16* vb = (__bf16*)(big + (size_t)NROWS * DIM * 4);
  __bf16* h1 = (__bf16*)(big);                   // reused after attention

  hipMemcpyAsync(x, x_in, (size_t)NROWS * DIM * sizeof(float),
                 hipMemcpyDeviceToDevice, stream);

  dim3 tb(32, 8);
  cvt_t_bf16<<<dim3(DIM / 32, DIM / 32),    tb, 0, stream>>>(Wq, Wq_t, DIM, DIM);
  cvt_t_bf16<<<dim3(DIM / 32, DIM / 32),    tb, 0, stream>>>(Wk, Wk_t, DIM, DIM);
  cvt_t_bf16<<<dim3(DIM / 32, DIM / 32),    tb, 0, stream>>>(Wv, Wv_t, DIM, DIM);
  cvt_t_bf16<<<dim3(MLPDIM / 32, DIM / 32), tb, 0, stream>>>(W1, W1_t, DIM, MLPDIM);
  cvt_t_bf16<<<dim3(DIM / 32, MLPDIM / 32), tb, 0, stream>>>(W2, W2_t, MLPDIM, DIM);

  dim3 gQKV(DIM / 128, NROWS / 128);
  dim3 gAtt(SEQ / 64, NHEADS, BATCH);
  dim3 gM1(MLPDIM / 128, NROWS / 128);
  dim3 gM2(DIM / 128, NROWS / 128);

  for (int layer = 0; layer < DEPTHL; ++layer) {
    layernorm_k<<<NROWS, 256, 0, stream>>>(x, gamma, beta, xn);
    gemm_k<0><<<gQKV, 256, 0, stream>>>(xn, Wq_t, bq, qb, nullptr, NROWS, DIM, DIM);
    gemm_k<0><<<gQKV, 256, 0, stream>>>(xn, Wk_t, bk, kb, nullptr, NROWS, DIM, DIM);
    gemm_k<0><<<gQKV, 256, 0, stream>>>(xn, Wv_t, bv, vb, nullptr, NROWS, DIM, DIM);
    attn_k<<<gAtt, 256, 0, stream>>>(qb, kb, vb, x);
    layernorm_k<<<NROWS, 256, 0, stream>>>(x, gamma, beta, xn);
    gemm_k<1><<<gM1, 256, 0, stream>>>(xn, W1_t, b1, h1, nullptr, NROWS, MLPDIM, DIM);
    gemm_k<2><<<gM2, 256, 0, stream>>>(h1, W2_t, b2, x, x, NROWS, DIM, MLPDIM);
  }
}
